// FastAttention_43731357007880
// MI455X (gfx1250) — compile-verified, matrix operand order checked by reading, not run on hardware
//
#include <hip/hip_runtime.h>
#include <hip/hip_bf16.h>

#define SEQ    4096
#define DH     64
#define NF     256
#define CHUNK  32
#define NCHUNK 128
#define CONDC  4      // 128 conditioning positions = 4 chunks
#define BHN    64     // B*H

typedef __attribute__((ext_vector_type(16))) __bf16 v16bf;
typedef __attribute__((ext_vector_type(8)))  __bf16 v8bf;
typedef __attribute__((ext_vector_type(8)))  float  v8f;

// native bf16 converts (gfx1250 has hardware bf16 cvt; RNE)
static __device__ __forceinline__ __bf16 f2bf(float f) { return (__bf16)f; }
static __device__ __forceinline__ float bf2f(__bf16 b) { return (float)b; }

// order-preserving float <-> uint encoding for atomicMax
static __device__ __forceinline__ unsigned encf(float f) {
  unsigned u = __float_as_uint(f);
  return u ^ ((u >> 31) ? 0xFFFFFFFFu : 0x80000000u);
}
static __device__ __forceinline__ float decf(unsigned u) {
  unsigned b = (u >> 31) ? (u ^ 0x80000000u) : ~u;
  return __uint_as_float(b);
}

// CDNA5 16x16x32 bf16 operand: per lane, K-elements are two contiguous 16B runs:
//   k = (lane>>4)*8 + e  (e=0..7)  and  k = (lane>>4)*8 + 16 + e  (e=8..15)
// => two aligned ds_load_b128 + register concat.
static __device__ __forceinline__ v16bf ld_op(const __bf16* __restrict__ buf,
                                              int j0, int ld, int k0, int lane) {
  const __bf16* p = buf + (j0 + (lane & 15)) * ld + k0 + (lane >> 4) * 8;
  v8bf lo = *(const v8bf*)(p);
  v8bf hi = *(const v8bf*)(p + 16);
  return __builtin_shufflevector(lo, hi, 0, 1, 2, 3, 4, 5, 6, 7,
                                 8, 9, 10, 11, 12, 13, 14, 15);
}

static __device__ __forceinline__ v8f zero_v8f() {
  v8f z;
#pragma unroll
  for (int i = 0; i < 8; ++i) z[i] = 0.f;
  return z;
}
static __device__ __forceinline__ v8f wmma_bf16(v16bf a, v16bf b, v8f c) {
  return __builtin_amdgcn_wmma_f32_16x16x32_bf16(false, a, false, b, (short)0, c, false, false);
}

// async global -> LDS copy (ASYNCcnt-tracked); inline asm for toolchain portability
static __device__ __forceinline__ void async_b128(unsigned lds_off, const float* gaddr) {
  asm volatile("global_load_async_to_lds_b128 %0, %1, off"
               :: "v"(lds_off), "v"(gaddr) : "memory");
}
static __device__ __forceinline__ void wait_async0() {
  asm volatile("s_wait_asynccnt 0x0" ::: "memory");
}

// ---------------------------------------------------------------------------
// Kernel 0: init global max accumulator (encoded -inf)
__global__ void init_gmax(unsigned* g) { if (threadIdx.x == 0) *g = 0x007FFFFFu; }

// Kernel 1: global max of k's projected scores (reference: m = max(data_dash_k))
__global__ __launch_bounds__(256) void kmax_kernel(const float* __restrict__ k,
                                                   const float* __restrict__ proj,
                                                   unsigned* __restrict__ gmax) {
  extern __shared__ char sm0[];
  __bf16* Pl  = (__bf16*)sm0;            // 256x64 bf16
  float*  red = (float*)(sm0 + 32768);   // 256 f32
  const int tid = threadIdx.x;
  for (int i = tid; i < NF * DH; i += 256) Pl[i] = f2bf(proj[i]);
  __syncthreads();

  size_t row = (size_t)blockIdx.x * 256 + tid;      // (b,h,n) flat row of k
  const float* kr = k + row * DH;
  float xr[DH];
#pragma unroll
  for (int d = 0; d < DH; ++d) xr[d] = kr[d];

  float lmax = -3.402823466e38f;
  for (int m = 0; m < NF; ++m) {
    float acc = 0.f;
    const __bf16* pm = Pl + m * DH;
#pragma unroll
    for (int d = 0; d < DH; ++d) acc = fmaf(xr[d], bf2f(pm[d]), acc);
    lmax = fmaxf(lmax, acc);
  }
  lmax *= 0.35355339059327373f;   // 64^(-1/4) normalizer (positive => commutes with max)

  red[tid] = lmax;
  __syncthreads();
  for (int s = 128; s > 0; s >>= 1) {
    if (tid < s) red[tid] = fmaxf(red[tid], red[tid + s]);
    __syncthreads();
  }
  if (tid == 0) atomicMax(gmax, encf(red[0]));
}

// ---------------------------------------------------------------------------
// Kernel 2: fused feature-map + chunked causal linear attention.
// One workgroup (8 wave32) per (b,h); S (256x64 f32) carried in LDS.
// q/k/v chunks are double-buffered via global_load_async_to_lds_b128.
__global__ __launch_bounds__(256) void perf_attn_kernel(
    const float* __restrict__ q, const float* __restrict__ k,
    const float* __restrict__ v, const float* __restrict__ proj,
    float* __restrict__ out, const unsigned* __restrict__ gmax_enc)
{
  extern __shared__ char smem[];
  __bf16* Pb    = (__bf16*)(smem);             // 256x64 bf16 projection   32768
  float*  S     = (float*)(smem + 32768);      // 256x64 f32 state         65536
  __bf16* St    = (__bf16*)(smem + 98304);     // 64x256 bf16 = S^T        32768
  float*  ksum  = (float*)(smem + 131072);     // 256 f32                   1024
  float*  ddL   = (float*)(smem + 132096);     // 32x256 f32 q-scores      32768
  __bf16* QfL   = (__bf16*)(smem + 164864);    // 32x256 bf16              16384
  __bf16* Kf    = (__bf16*)(smem + 181248);    // 32x256 bf16              16384
  __bf16* KfT   = (__bf16*)(smem + 197632);    // 256x32 bf16 = Kf^T       16384
  __bf16* Xq    = (__bf16*)(smem + 214016);    // 32x64 bf16                4096
  __bf16* Xk    = (__bf16*)(smem + 218112);    // 32x64 bf16                4096
  __bf16* Vt    = (__bf16*)(smem + 222208);    // 64x32 bf16 = V^T          4096
  float*  AL    = (float*)(smem + 226304);     // 32x32 f32                 4096
  __bf16* Ab    = (__bf16*)(smem + 230400);    // 32x32 bf16                2048
  float*  red   = (float*)(smem + 232448);     // 2 x 256 f32 partials      2048
  float*  diagq = (float*)(smem + 234496);     // scalars (4x32 f32)
  float*  diagk = diagq + 32;
  float*  rmax  = diagq + 64;
  float*  Dinv  = diagq + 96;
  // double-buffered raw f32 staging (2 x 8KB each), addressed by byte offset
  const int STGQ = 235520, STGK = 251904, STGV = 268288;   // total = 284672 B
  auto stage = [&](int off, int buf) -> float* {
    return (float*)(smem + off + buf * (CHUNK * DH * 4));
  };

  const int tid  = threadIdx.x;
  const int lane = tid & 31;
  const int wave = tid >> 5;
  const size_t base = (size_t)blockIdx.x * SEQ * DH;
  const float NORM  = 0.35355339059327373f;   // 64^-0.25
  const float RATIO = 0.0625f;                // 256^-0.5
  const float gmax  = decf(*gmax_enc);

  for (int i = tid; i < NF * DH; i += 256) { S[i] = 0.f; Pb[i] = f2bf(proj[i]); }
  for (int i = tid; i < NF; i += 256) ksum[i] = 0.f;
  __syncthreads();

  // ---- issue async copy of one 32x64 f32 chunk into an LDS staging buffer --
  auto issue_chunk = [&](const float* src, int c, int stg_off, int buf) {
    const float* gp = src + base + (size_t)c * CHUNK * DH;
    unsigned ldsb = (unsigned)(size_t)stage(stg_off, buf);
#pragma unroll
    for (int it = 0; it < 2; ++it) {
      int idx = it * 256 + tid;                 // b128 index (512 per chunk)
      async_b128(ldsb + idx * 16, gp + idx * 4);
    }
  };
  // ---- staged f32 -> bf16*norm + ||x||^2 diag ------------------------------
  auto conv_x = [&](const float* stg, __bf16* dst, float* diag) {
    if (tid < CHUNK) diag[tid] = 0.f;
    __syncthreads();
    for (int i = tid; i < CHUNK * DH; i += 256) {
      float x = stg[i];
      dst[i] = f2bf(x * NORM);
      atomicAdd(&diag[i >> 6], x * x);
    }
    __syncthreads();
    if (tid < CHUNK) diag[tid] *= 0.0625f;  // 0.5 * 64^-0.5
  };
  // ---- staged V -> transposed bf16 (64x32) so it is always a row operand ---
  auto conv_v = [&](const float* stg) {
    for (int i = tid; i < CHUNK * DH; i += 256)
      Vt[(i & 63) * CHUNK + (i >> 6)] = f2bf(stg[i]);
  };

  // ---- k features: GEMM fused with exp transform (global max, no row red) --
  auto feat_k = [&]() {
#pragma unroll
    for (int i = 0; i < 4; ++i) {
      int t = wave * 4 + i;               // 32 tiles over 8 waves
      int mt = t >> 4, nt = t & 15;
      v8f acc = zero_v8f();
      acc = wmma_bf16(ld_op(Xk, mt * 16, DH, 0,  lane), ld_op(Pb, nt * 16, DH, 0,  lane), acc);
      acc = wmma_bf16(ld_op(Xk, mt * 16, DH, 32, lane), ld_op(Pb, nt * 16, DH, 32, lane), acc);
#pragma unroll
      for (int r = 0; r < 8; ++r) {
        int m = mt * 16 + r + (lane >> 4) * 8;      // chunk position
        int n = nt * 16 + (lane & 15);              // feature
        __bf16 b = f2bf(RATIO * (__expf(acc[r] - diagk[m] - gmax) + 1e-4f));
        Kf[m * NF + n]     = b;
        KfT[n * CHUNK + m] = b;
      }
    }
  };

  // ---- q scores GEMM -> ddL (row max still needed) -------------------------
  auto feat_q = [&]() {
#pragma unroll
    for (int i = 0; i < 4; ++i) {
      int t = wave * 4 + i;
      int mt = t >> 4, nt = t & 15;
      v8f acc = zero_v8f();
      acc = wmma_bf16(ld_op(Xq, mt * 16, DH, 0,  lane), ld_op(Pb, nt * 16, DH, 0,  lane), acc);
      acc = wmma_bf16(ld_op(Xq, mt * 16, DH, 32, lane), ld_op(Pb, nt * 16, DH, 32, lane), acc);
#pragma unroll
      for (int r = 0; r < 8; ++r)
        ddL[(mt * 16 + r + (lane >> 4) * 8) * NF + nt * 16 + (lane & 15)] = acc[r];
    }
  };

  // ---- row max of ddL: 8 threads per row, then combine ---------------------
  auto rowmax_partial = [&]() {
    int row = tid >> 3, seg = tid & 7;
    const float* p = ddL + row * NF + seg * 32;
    float mx = -3.402823466e38f;
    for (int n = 0; n < 32; ++n) mx = fmaxf(mx, p[n]);
    red[tid] = mx;
  };
  auto rowmax_combine = [&]() {
    if (tid < CHUNK) {
      float mx = red[tid * 8];
#pragma unroll
      for (int j = 1; j < 8; ++j) mx = fmaxf(mx, red[tid * 8 + j]);
      rmax[tid] = mx;
    }
  };

  auto transform_q = [&]() {
    for (int i = tid; i < CHUNK * NF; i += 256) {
      int r = i >> 8;
      QfL[i] = f2bf(RATIO * (__expf(ddL[i] - diagq[r] - rmax[r]) + 1e-4f));
    }
  };

  // ---- S(256x64) += Kf^T(256x32) @ V(32x64); ksum += colsum(Kf) ------------
  auto s_update = [&]() {
#pragma unroll
    for (int i = 0; i < 8; ++i) {
      int t = wave + i * 8;               // 64 tiles over 8 waves
      int mt = t >> 2, nt = t & 3;
      v8f acc;
#pragma unroll
      for (int r = 0; r < 8; ++r)
        acc[r] = S[(mt * 16 + r + (lane >> 4) * 8) * DH + nt * 16 + (lane & 15)];
      acc = wmma_bf16(ld_op(KfT, mt * 16, CHUNK, 0, lane),
                      ld_op(Vt,  nt * 16, CHUNK, 0, lane), acc);
#pragma unroll
      for (int r = 0; r < 8; ++r)
        S[(mt * 16 + r + (lane >> 4) * 8) * DH + nt * 16 + (lane & 15)] = acc[r];
    }
    {
      const __bf16* p = KfT + tid * CHUNK;          // feature row, contiguous
      float s = 0.f;
      for (int kk = 0; kk < CHUNK; ++kk) s += bf2f(p[kk]);
      ksum[tid] += s;
    }
  };

  // ---- refresh bf16 transposed state shadow --------------------------------
  auto st_refresh = [&]() {
    for (int i = tid; i < NF * DH; i += 256)
      St[(i & 63) * NF + (i >> 6)] = f2bf(S[i]);
  };

  // ---- out = (Qf @ S [+ A_masked @ V]) * Dinv ------------------------------
  auto numer_store = [&](int c, bool withA) {
    int mt = wave >> 2, nt = wave & 3;    // 8 tiles over 8 waves
    v8f a0 = zero_v8f(), a1 = zero_v8f();
#pragma unroll
    for (int ks = 0; ks < NF; ks += 64) { // 2 interleaved accumulators
      a0 = wmma_bf16(ld_op(QfL, mt * 16, NF, ks,      lane),
                     ld_op(St,  nt * 16, NF, ks,      lane), a0);
      a1 = wmma_bf16(ld_op(QfL, mt * 16, NF, ks + 32, lane),
                     ld_op(St,  nt * 16, NF, ks + 32, lane), a1);
    }
#pragma unroll
    for (int r = 0; r < 8; ++r) a0[r] += a1[r];
    if (withA)
      a0 = wmma_bf16(ld_op(Ab, mt * 16, CHUNK, 0, lane),
                     ld_op(Vt, nt * 16, CHUNK, 0, lane), a0);
    float* og = out + base + (size_t)c * CHUNK * DH;
#pragma unroll
    for (int r = 0; r < 8; ++r) {
      int m = mt * 16 + r + (lane >> 4) * 8;
      og[m * DH + nt * 16 + (lane & 15)] = a0[r] * Dinv[m];
    }
  };

  // ================= Phase 1: conditioning block, accumulate S & ksum =======
  issue_chunk(k, 0, STGK, 0);
  issue_chunk(v, 0, STGV, 0);
  for (int c = 0; c < CONDC; ++c) {
    wait_async0();
    __syncthreads();
    if (c + 1 < CONDC) { issue_chunk(k, c + 1, STGK, (c + 1) & 1);
                         issue_chunk(v, c + 1, STGV, (c + 1) & 1); }
    conv_x(stage(STGK, c & 1), Xk, diagk);
    conv_v(stage(STGV, c & 1));
    __syncthreads();
    feat_k();
    __syncthreads();
    s_update();
    __syncthreads();
  }
  st_refresh();
  __syncthreads();

  // ================= Phase 2: conditioning outputs (non-causal, no eps) =====
  issue_chunk(q, 0, STGQ, 0);
  for (int c = 0; c < CONDC; ++c) {
    wait_async0();
    __syncthreads();
    if (c + 1 < CONDC) issue_chunk(q, c + 1, STGQ, (c + 1) & 1);
    conv_x(stage(STGQ, c & 1), Xq, diagq);
    __syncthreads();
    feat_q();
    __syncthreads();
    rowmax_partial();
    __syncthreads();
    rowmax_combine();
    __syncthreads();
    transform_q();
    __syncthreads();
    {   // qf . cond_k_sum, 8 threads/row partials
      int row = tid >> 3, seg = tid & 7;
      const __bf16* p = QfL + row * NF + seg * 32;
      const float*  kp = ksum + seg * 32;
      float dp = 0.f;
      for (int n = 0; n < 32; ++n) dp += bf2f(p[n]) * kp[n];
      red[tid] = dp;
    }
    __syncthreads();
    if (tid < CHUNK) {
      float dp = red[tid * 8];
#pragma unroll
      for (int j = 1; j < 8; ++j) dp += red[tid * 8 + j];
      Dinv[tid] = 1.f / dp;
    }
    __syncthreads();
    numer_store(c, false);
    __syncthreads();
  }

  // ================= Phase 3: causal chunks =================================
  issue_chunk(k, CONDC, STGK, 0);
  issue_chunk(v, CONDC, STGV, 0);
  issue_chunk(q, CONDC, STGQ, 0);
  for (int c = CONDC; c < NCHUNK; ++c) {
    wait_async0();
    __syncthreads();
    if (c + 1 < NCHUNK) {
      issue_chunk(k, c + 1, STGK, (c + 1) & 1);
      issue_chunk(v, c + 1, STGV, (c + 1) & 1);
      issue_chunk(q, c + 1, STGQ, (c + 1) & 1);
    }
    conv_x(stage(STGK, c & 1), Xk, diagk);
    conv_v(stage(STGV, c & 1));
    conv_x(stage(STGQ, c & 1), Xq, diagq);
    __syncthreads();

    feat_k();          // writes Kf/KfT
    feat_q();          // writes ddL (disjoint)
    __syncthreads();
    rowmax_partial();
    __syncthreads();
    rowmax_combine();
    __syncthreads();
    transform_q();
    __syncthreads();

    // A = causal_mask(Qf @ Kf^T)  (4 tiles on waves 0..3; wave-uniform branch)
    if (wave < 4) {
      int mt = wave >> 1, nt = wave & 1;
      v8f a0 = zero_v8f(), a1 = zero_v8f();
#pragma unroll
      for (int ks = 0; ks < NF; ks += 64) {
        a0 = wmma_bf16(ld_op(QfL, mt * 16, NF, ks,      lane),
                       ld_op(Kf,  nt * 16, NF, ks,      lane), a0);
        a1 = wmma_bf16(ld_op(QfL, mt * 16, NF, ks + 32, lane),
                       ld_op(Kf,  nt * 16, NF, ks + 32, lane), a1);
      }
#pragma unroll
      for (int r = 0; r < 8; ++r) {
        int m = mt * 16 + r + (lane >> 4) * 8;
        int j = nt * 16 + (lane & 15);
        float val = (j <= m) ? (a0[r] + a1[r]) : 0.f;
        AL[m * CHUNK + j] = val;
        Ab[m * CHUNK + j] = f2bf(val);
      }
    }
    {   // partials: rowsum(qf) and qf.k_prev (pre-update ksum)
      int row = tid >> 3, seg = tid & 7;
      const __bf16* p = QfL + row * NF + seg * 32;
      const float*  kp = ksum + seg * 32;
      float s = 0.f, dp = 0.f;
      for (int n = 0; n < 32; ++n) { float qv = bf2f(p[n]); s += qv; dp += qv * kp[n]; }
      red[tid] = s; red[256 + tid] = dp;
    }
    __syncthreads();
    if (tid < CHUNK) {   // D = qf.k_prev + rowsum(A_masked) + eps*rowsum(qf)
      float s = 0.f, dp = 0.f;
#pragma unroll
      for (int j = 0; j < 8; ++j) { s += red[tid * 8 + j]; dp += red[256 + tid * 8 + j]; }
      float sa = 0.f;
      for (int j = 0; j < CHUNK; ++j) sa += AL[tid * CHUNK + j];
      Dinv[tid] = 1.f / (dp + sa + 1e-6f * s);
    }
    __syncthreads();
    numer_store(c, true);
    __syncthreads();
    s_update();
    __syncthreads();
    st_refresh();
    __syncthreads();
  }
}

// ---------------------------------------------------------------------------
extern "C" void kernel_launch(void* const* d_in, const int* in_sizes, int n_in,
                              void* d_out, int out_size, void* d_ws, size_t ws_size,
                              hipStream_t stream) {
  (void)in_sizes; (void)n_in; (void)out_size; (void)ws_size;
  const float* q    = (const float*)d_in[0];
  const float* k    = (const float*)d_in[1];
  const float* v    = (const float*)d_in[2];
  const float* proj = (const float*)d_in[3];
  float* out = (float*)d_out;
  unsigned* gmax = (unsigned*)d_ws;

  (void)hipFuncSetAttribute(reinterpret_cast<const void*>(kmax_kernel),
                            hipFuncAttributeMaxDynamicSharedMemorySize, 33792);
  (void)hipFuncSetAttribute(reinterpret_cast<const void*>(perf_attn_kernel),
                            hipFuncAttributeMaxDynamicSharedMemorySize, 286720);

  init_gmax<<<1, 64, 0, stream>>>(gmax);
  kmax_kernel<<<(4 * 16 * SEQ) / 256, 256, 33792, stream>>>(k, proj, gmax);
  perf_attn_kernel<<<BHN, 256, 284672, stream>>>(q, k, v, proj, out, gmax);
}
